// EfficientDetTransformerFastRCNN_87256555585678
// MI455X (gfx1250) — matrix operand order repaired and checked by reading, not compiled, not run on hardware
//
#include <hip/hip_runtime.h>
#include <stdint.h>
#include <stddef.h>

// ---------------------------------------------------------------------------
// CDNA5 (gfx1250): transformer stack on p4 only (outputs depend solely on
// enh[1]); all GEMMs via v_wmma_f32_16x16x32_bf16 (f32 accum); A and B tiles
// staged to LDS by the Tensor Data Mover with DOUBLE BUFFERING so the DMA of
// tile t+1 overlaps the WMMAs of tile t; b128 LDS fragment loads; fused
// epilogues (alpha/bias/relu).  GEMM B operand is column-major [N][K] so the
// TDM can stage it and weight packing is a pure dtype cast.
// ---------------------------------------------------------------------------

typedef __attribute__((ext_vector_type(16))) __bf16 v16bf;
typedef __attribute__((ext_vector_type(8)))  __bf16 v8bf;
typedef __attribute__((ext_vector_type(8)))  float  v8f;
typedef __attribute__((ext_vector_type(4)))  unsigned int u32x4;
typedef __attribute__((ext_vector_type(8)))  int    i32x8;
typedef __attribute__((ext_vector_type(4)))  int    i32x4;

#if defined(__HIP_DEVICE_COMPILE__)
# if __has_builtin(__builtin_amdgcn_tensor_load_to_lds) && \
     __has_builtin(__builtin_amdgcn_s_wait_tensorcnt)
#  define EDT_HAVE_TDM 1
# endif
#endif
#ifndef EDT_HAVE_TDM
#define EDT_HAVE_TDM 0
#endif

#define BM 64
#define BN 64
#define BK 32
#define LDSW 40   // 80B LDS row stride: banks-friendly, matches TDM pad

// Issue one TDM 2D tile load: tileRows x 32 bf16 elems, row stride ldK elems,
// zero-fill outside (kRem x mRem); LDS rows padded to 80 bytes.
__device__ __forceinline__ void edt_tdm_tile(const __bf16* gsrc, unsigned lds_off,
                                             int kRem, int mRem, int ldK, int tileRows) {
#if EDT_HAVE_TDM
  unsigned long long ga = (unsigned long long)(size_t)(const void*)gsrc;
  u32x4 g0 = {};
  g0[0] = 1u;                                   // one descriptor
  g0[1] = lds_off;                              // LDS destination byte offset
  g0[2] = (unsigned)(ga & 0xffffffffu);         // global_addr[31:0]
  g0[3] = (unsigned)((ga >> 32) & 0x01ffffffu) | (2u << 30);   // addr[56:32], type=2
  i32x8 g1 = {};
  g1[0] = (int)((1u << 16)      // data_size = 2 bytes
              | (1u << 20)      // pad_enable
              | (3u << 22)      // pad_interval = 16 dwords (one 64B tile row)
              | (3u << 25));    // pad_amount   = 4 dwords  -> 80B LDS row stride
  unsigned td0 = (unsigned)kRem, td1 = (unsigned)mRem;
  g1[1] = (int)((td0 & 0xffffu) << 16);                  // tensor_dim0 lo
  g1[2] = (int)((td0 >> 16) | ((td1 & 0xffffu) << 16));  // dim0 hi | dim1 lo
  g1[3] = (int)((td1 >> 16) | (32u << 16));              // dim1 hi | tile_dim0 = 32
  g1[4] = tileRows & 0xffff;                             // tile_dim1, tile_dim2 = 0
  g1[5] = (int)(unsigned)ldK;                            // tensor_dim0_stride
  i32x4 gz = {};
# if __clang_major__ >= 23
  i32x8 gz8 = {};
  __builtin_amdgcn_tensor_load_to_lds(g0, g1, gz, gz, gz8, 0);
# else
  __builtin_amdgcn_tensor_load_to_lds(g0, g1, gz, gz, 0);
# endif
#else
  (void)gsrc; (void)lds_off; (void)kRem; (void)mRem; (void)ldK; (void)tileRows;
#endif
}

// ---------------------------------------------------------------------------
// Batched GEMM  D[b] = relu( alpha * A[b] x B[b] + bias )
//   A: [M,K] bf16 row-major (ld=K);  B: column-major [N][K] bf16 (ld=K)
//   D: [M,N] f32 row-major.  Block = 128 thr = 4 waves, 64x64 tile,
//   each wave computes 32x32 via 4 WMMAs.  TDM double-buffers the LDS tiles.
// ---------------------------------------------------------------------------
__global__ __launch_bounds__(128)
void edt_gemm_bf16(const __bf16* __restrict__ A, const __bf16* __restrict__ Bm,
                   const float* __restrict__ bias, float* __restrict__ D,
                   int M, int N, int K, float alpha, int relu,
                   long long sA, long long sB, long long sD)
{
  __shared__ __bf16 As[2][BM][LDSW];   // As[buf][m][k]
  __shared__ __bf16 Bs[2][BN][LDSW];   // Bs[buf][n][k]   (B column-major)

  const int tid  = threadIdx.x;
  const int lane = tid & 31;
  const int wave = tid >> 5;
  const int rowBase = blockIdx.y * BM;
  const int colBase = blockIdx.x * BN;
  A  += (long long)blockIdx.z * sA;
  Bm += (long long)blockIdx.z * sB;
  D  += (long long)blockIdx.z * sD;

  const int tm = (wave >> 1) * 32;     // wave's 32x32 tile inside 64x64
  const int tn = (wave & 1) * 32;
  const int khalf = lane >> 4;
  const int l15   = lane & 15;

  v8f zero = {};
  v8f acc00 = zero, acc01 = zero, acc10 = zero, acc11 = zero;

  const int nk = (K + BK - 1) / BK;

#if EDT_HAVE_TDM
  if (wave == 0) {   // prologue: tile 0 into buffer 0
    edt_tdm_tile(A,  (unsigned)(size_t)(void*)&As[0][0][0], K, M - rowBase, K, BM);
    edt_tdm_tile(Bm + (long long)colBase * K,
                 (unsigned)(size_t)(void*)&Bs[0][0][0], K, N - colBase, K, BN);
  }
#else
  const bool alignedK = ((K & 7) == 0);
#endif

  for (int t = 0; t < nk; ++t) {
    const int buf = t & 1;
    const int k0 = t * BK;
#if EDT_HAVE_TDM
    if (wave == 0)
      __builtin_amdgcn_s_wait_tensorcnt(0);     // tile t has landed
    __syncthreads();                            // also: prior readers of other buf done
    if (wave == 0 && t + 1 < nk) {              // issue tile t+1 while computing tile t
      const int k1 = k0 + BK;
      edt_tdm_tile(A  + (long long)rowBase * K + k1,
                   (unsigned)(size_t)(void*)&As[buf ^ 1][0][0], K - k1, M - rowBase, K, BM);
      edt_tdm_tile(Bm + (long long)colBase * K + k1,
                   (unsigned)(size_t)(void*)&Bs[buf ^ 1][0][0], K - k1, N - colBase, K, BN);
    }
#else
    // manual staging: 2 chunks of 8 contiguous K-elems per thread per matrix
    if (t) __syncthreads();                     // previous tile reads complete
    {
      const bool fullK = (k0 + BK <= K) && alignedK;
#pragma unroll
      for (int c = 0; c < 2; ++c) {
        int linear = tid + 128 * c;
        int r = linear >> 2, kseg = (linear & 3) * 8;
        // A tile
        if (fullK && (rowBase + BM <= M)) {
          *(v8bf*)&As[buf][r][kseg] = *(const v8bf*)(A + (long long)(rowBase + r) * K + k0 + kseg);
        } else {
#pragma unroll
          for (int e = 0; e < 8; ++e) {
            int gr = rowBase + r, gc = k0 + kseg + e;
            As[buf][r][kseg + e] = (gr < M && gc < K) ? A[(long long)gr * K + gc] : (__bf16)0.0f;
          }
        }
        // B tile
        if (fullK && (colBase + BN <= N)) {
          *(v8bf*)&Bs[buf][r][kseg] = *(const v8bf*)(Bm + (long long)(colBase + r) * K + k0 + kseg);
        } else {
#pragma unroll
          for (int e = 0; e < 8; ++e) {
            int gn = colBase + r, gc = k0 + kseg + e;
            Bs[buf][r][kseg + e] = (gn < N && gc < K) ? Bm[(long long)gn * K + gc] : (__bf16)0.0f;
          }
        }
      }
    }
    __syncthreads();
#endif
    if (k0 + BK < K)                   // warm GL2 for the next K slice
      __builtin_prefetch(A + (long long)(rowBase + (tid >> 1)) * K + k0 + BK, 0, 1);

    // fragments: each is two contiguous 16B LDS chunks (ds_load_b128)
    auto ldA = [&](int mrow) -> v16bf {
      const __bf16* p = &As[buf][mrow][0];
      v8bf lo = *(const v8bf*)(p + 8 * khalf);
      v8bf hi = *(const v8bf*)(p + 16 + 8 * khalf);
      return __builtin_shufflevector(lo, hi, 0,1,2,3,4,5,6,7,8,9,10,11,12,13,14,15);
    };
    auto ldB = [&](int ncol) -> v16bf {
      const __bf16* p = &Bs[buf][ncol][0];
      v8bf lo = *(const v8bf*)(p + 8 * khalf);
      v8bf hi = *(const v8bf*)(p + 16 + 8 * khalf);
      return __builtin_shufflevector(lo, hi, 0,1,2,3,4,5,6,7,8,9,10,11,12,13,14,15);
    };
    v16bf a0 = ldA(tm + l15), a1 = ldA(tm + 16 + l15);
    v16bf b0 = ldB(tn + l15), b1 = ldB(tn + 16 + l15);

    acc00 = __builtin_amdgcn_wmma_f32_16x16x32_bf16(false, a0, false, b0, (short)0, acc00, false, false);
    acc01 = __builtin_amdgcn_wmma_f32_16x16x32_bf16(false, a0, false, b1, (short)0, acc01, false, false);
    acc10 = __builtin_amdgcn_wmma_f32_16x16x32_bf16(false, a1, false, b0, (short)0, acc10, false, false);
    acc11 = __builtin_amdgcn_wmma_f32_16x16x32_bf16(false, a1, false, b1, (short)0, acc11, false, false);
  }

  // fused epilogue
  auto epi = [&](v8f acc, int mt, int nt) {
#pragma unroll
    for (int r = 0; r < 8; ++r) {
      int row = rowBase + tm + mt + khalf * 8 + r;    // C/D layout: VGPR r -> M=r / 8+r
      int col = colBase + tn + nt + l15;
      if (row < M && col < N) {
        float v = acc[r] * alpha;
        if (bias) v += bias[col];
        if (relu) v = fmaxf(v, 0.0f);
        D[(long long)row * N + col] = v;
      }
    }
  };
  epi(acc00, 0, 0); epi(acc01, 0, 16); epi(acc10, 16, 0); epi(acc11, 16, 16);
}

// ---------------------------------------------------------------------------
// Elementwise / pack / reshape helpers
// ---------------------------------------------------------------------------
__global__ void edt_pack_rows(const float* __restrict__ x, __bf16* __restrict__ y, int n) {
  int i = blockIdx.x * blockDim.x + threadIdx.x;
  if (i < n) y[i] = (__bf16)x[i];
}

// [B,C,HW] f32 -> rows[(b*HW+hw), c] bf16
__global__ void edt_nchw_to_rows(const float* __restrict__ x, __bf16* __restrict__ rows,
                                 int Bn, int Cn, int HW) {
  int i = blockIdx.x * blockDim.x + threadIdx.x;
  if (i >= Bn * HW * Cn) return;
  int c = i % Cn; int t = i / Cn; int hw = t % HW; int b = t / HW;
  rows[i] = (__bf16)x[((long long)b * Cn + c) * HW + hw];
}

// enh = feat + rows (rows in (b,hw),c layout) -> NCHW f32
__global__ void edt_add_residual(const float* __restrict__ feat, const float* __restrict__ yr,
                                 float* __restrict__ enh, int Bn, int Cn, int HW) {
  int i = blockIdx.x * blockDim.x + threadIdx.x;
  if (i >= Bn * Cn * HW) return;
  int hw = i % HW; int t = i / HW; int c = t % Cn; int b = t / Cn;
  enh[i] = feat[i] + yr[((long long)b * HW + hw) * Cn + c];
}

// rows[(b*HW+hw), o] f32 -> out[b,o,hw] f32
__global__ void edt_rows_to_nchw(const float* __restrict__ yr, float* __restrict__ outp,
                                 int Bn, int O, int HW) {
  int i = blockIdx.x * blockDim.x + threadIdx.x;
  if (i >= Bn * O * HW) return;
  int hw = i % HW; int t = i / HW; int o = t % O; int b = t / O;
  outp[i] = yr[((long long)b * HW + hw) * O + o];
}

// qkv rows [(b*S+s),336] -> Qp/Kp [bh,S,16] (A rowmaj / B colmaj), Vt [bh,16,S]
__global__ void edt_pack_qkv(const float* __restrict__ qkv, __bf16* __restrict__ Qp,
                             __bf16* __restrict__ Kp, __bf16* __restrict__ Vt, int S) {
  int i = blockIdx.x * blockDim.x + threadIdx.x;
  int total = 2 * 8 * S * 16;
  if (i >= total) return;
  int d = i & 15; int t = i >> 4;
  int s = t % S; t /= S;
  int h = t % 8; int b = t / 8;
  float q = 0.f, k = 0.f, v = 0.f;
  if (d < 14) {
    const float* base = qkv + ((long long)b * S + s) * 336;
    q = base[h * 14 + d];
    k = base[112 + h * 14 + d];
    v = base[224 + h * 14 + d];
  }
  long long bh = (long long)b * 8 + h;
  Qp[(bh * S + s) * 16 + d] = (__bf16)q;
  Kp[(bh * S + s) * 16 + d] = (__bf16)k;   // column-major B of K^T == K as stored
  Vt[(bh * 16 + d) * S + s] = (__bf16)v;   // column-major B of V
}

// O[bh,S,16] f32 -> rows[(b*S+s), h*14+d] bf16
__global__ void edt_merge_heads(const float* __restrict__ O, __bf16* __restrict__ out, int S) {
  int i = blockIdx.x * blockDim.x + threadIdx.x;
  int total = 2 * 8 * S * 14;
  if (i >= total) return;
  int d = i % 14; int t = i / 14;
  int s = t % S; t /= S;
  int h = t % 8; int b = t / 8;
  out[((long long)b * S + s) * 112 + h * 14 + d] =
      (__bf16)O[(((long long)b * 8 + h) * S + s) * 16 + d];
}

// one wave per row: softmax, emit bf16 directly
__global__ void edt_softmax(const float* __restrict__ x, __bf16* __restrict__ y, int R, int S) {
  int row = blockIdx.x * (blockDim.x >> 5) + (threadIdx.x >> 5);
  if (row >= R) return;
  int lane = threadIdx.x & 31;
  const float* xr = x + (long long)row * S;
  float mx = -3.402823466e38f;
  for (int c = lane; c < S; c += 32) mx = fmaxf(mx, xr[c]);
#pragma unroll
  for (int o = 16; o > 0; o >>= 1) mx = fmaxf(mx, __shfl_xor(mx, o, 32));
  float sum = 0.f;
  for (int c = lane; c < S; c += 32) sum += __expf(xr[c] - mx);
#pragma unroll
  for (int o = 16; o > 0; o >>= 1) sum += __shfl_xor(sum, o, 32);
  float inv = 1.0f / sum;
  __bf16* yr = y + (long long)row * S;
  for (int c = lane; c < S; c += 32) yr[c] = (__bf16)(__expf(xr[c] - mx) * inv);
}

// one wave per row: y = LN(x + d) * g + b ; emits f32 and bf16 repack
__global__ void edt_ln(const float* __restrict__ x, const float* __restrict__ dlt,
                       const float* __restrict__ g, const float* __restrict__ b,
                       float* __restrict__ yf, __bf16* __restrict__ yb, int M, int C) {
  int row = blockIdx.x * (blockDim.x >> 5) + (threadIdx.x >> 5);
  if (row >= M) return;
  int lane = threadIdx.x & 31;
  const float* xr = x + (long long)row * C;
  const float* dr = dlt + (long long)row * C;
  float s = 0.f;
  for (int c = lane; c < C; c += 32) s += xr[c] + dr[c];
#pragma unroll
  for (int o = 16; o > 0; o >>= 1) s += __shfl_xor(s, o, 32);
  float mean = s / (float)C;
  float vs = 0.f;
  for (int c = lane; c < C; c += 32) { float t = xr[c] + dr[c] - mean; vs += t * t; }
#pragma unroll
  for (int o = 16; o > 0; o >>= 1) vs += __shfl_xor(vs, o, 32);
  float inv = rsqrtf(vs / (float)C + 1e-5f);
  for (int c = lane; c < C; c += 32) {
    float t = (xr[c] + dr[c] - mean) * inv * g[c] + b[c];
    yf[(long long)row * C + c] = t;
    yb[(long long)row * C + c] = (__bf16)t;
  }
}

// im2col for 3x3 pad-1 conv: x[B,C,H,W] f32 -> col[(b*HW+hw), c*9+ky*3+kx] bf16
__global__ void edt_im2col(const float* __restrict__ x, __bf16* __restrict__ col,
                           int Bn, int Cn, int H, int W) {
  int i = blockIdx.x * blockDim.x + threadIdx.x;
  int CK = Cn * 9;
  if (i >= Bn * H * W * CK) return;
  int cc = i % CK; int row = i / CK;
  int c = cc / 9; int k = cc % 9; int ky = k / 3; int kx = k % 3;
  int b = row / (H * W); int hw = row % (H * W);
  int y = hw / W; int xx = hw % W;
  int iy = y + ky - 1, ix = xx + kx - 1;
  float v = 0.f;
  if (iy >= 0 && iy < H && ix >= 0 && ix < W)
    v = x[(((long long)b * Cn + c) * H + iy) * W + ix];
  col[i] = (__bf16)v;
}

// RoIPool(7x7 adaptive max), one block per ROI
__global__ void edt_roipool(const float* __restrict__ feat, const float* __restrict__ rois,
                            float* __restrict__ out, int C, int H, int W) {
  int n = blockIdx.x;
  const float* r = rois + (long long)n * 5;
  int bidx = (int)r[0];
  int bx1 = (int)roundf(r[1] * 0.0625f);
  int by1 = (int)roundf(r[2] * 0.0625f);
  int bx2 = (int)roundf(r[3] * 0.0625f);
  int by2 = (int)roundf(r[4] * 0.0625f);
  int x1 = bx1 > 0 ? bx1 : 0;
  int y1 = by1 > 0 ? by1 : 0;
  int x2 = bx2 < W - 1 ? bx2 : W - 1;
  int y2 = by2 < H - 1 ? by2 : H - 1;
  int h = y2 - y1 + 1, w = x2 - x1 + 1;
  for (int idx = threadIdx.x; idx < C * 49; idx += blockDim.x) {
    int c = idx / 49; int j = (idx % 49) / 7; int i2 = idx % 7;
    int ys = y1 + (j * h) / 7,  ye = y1 + ((j + 1) * h + 6) / 7;
    int xs = x1 + (i2 * w) / 7, xe = x1 + ((i2 + 1) * w + 6) / 7;
    float m = -3.402823466e38f;
    for (int yy = ys; yy < ye; ++yy)
      for (int xx = xs; xx < xe; ++xx)
        m = fmaxf(m, feat[(((long long)bidx * C + c) * H + yy) * W + xx]);
    out[(long long)n * C * 49 + idx] = m;
  }
}

// ---------------------------------------------------------------------------
// Host orchestration
// ---------------------------------------------------------------------------
static inline int cdiv_i(int a, int b) { return (a + b - 1) / b; }

extern "C" void kernel_launch(void* const* d_in, const int* in_sizes, int n_in,
                              void* d_out, int out_size, void* d_ws, size_t ws_size,
                              hipStream_t stream) {
  (void)in_sizes; (void)n_in; (void)out_size; (void)ws_size;

  const int C = 112, Bn = 2, H4 = 24, W4 = 24;
  const int HW = H4 * W4;            // 576
  const int S = HW, M = Bn * HW;     // 576, 1152
  const int NROI = 512, NC = 81;
  const float ATT_SCALE = 0.2672612419f;   // 1/sqrt(14)

  // --- input map (setup_inputs dict flattening order) ---
  const float* feat4     = (const float*)d_in[1];       // p4
  const float* proposals = (const float*)d_in[5];
  const int stBase = 6 + 1 * 28;                        // st_params[1]
  const float* pin_w  = (const float*)d_in[stBase + 0];
  const float* pin_b  = (const float*)d_in[stBase + 1];
  const float* pout_w = (const float*)d_in[stBase + 2];
  const float* pout_b = (const float*)d_in[stBase + 3];
  const float* Lw[2][12];
  for (int l = 0; l < 2; ++l)
    for (int k = 0; k < 12; ++k)
      Lw[l][k] = (const float*)d_in[stBase + 4 + l * 12 + k];
  const int rpnBase = 6 + 5 * 28;                       // 146
  const float* c1_w   = (const float*)d_in[rpnBase + 0];
  const float* c1_b   = (const float*)d_in[rpnBase + 1];
  const float* rcls_w = (const float*)d_in[rpnBase + 2];
  const float* rcls_b = (const float*)d_in[rpnBase + 3];
  const float* rbb_w  = (const float*)d_in[rpnBase + 4];
  const float* rbb_b  = (const float*)d_in[rpnBase + 5];
  const int hdBase = rpnBase + 6;                       // 152
  const float* fc1_w = (const float*)d_in[hdBase + 0];
  const float* fc1_b = (const float*)d_in[hdBase + 1];
  const float* fc2_w = (const float*)d_in[hdBase + 2];
  const float* fc2_b = (const float*)d_in[hdBase + 3];
  const float* hcls_w = (const float*)d_in[hdBase + 4];
  const float* hcls_b = (const float*)d_in[hdBase + 5];
  const float* hbb_w  = (const float*)d_in[hdBase + 6];
  const float* hbb_b  = (const float*)d_in[hdBase + 7];

  float* out = (float*)d_out;
  const long long O_CLS = 0;
  const long long O_BBOX = (long long)NROI * NC;                  // 41472
  const long long O_RPNC = O_BBOX + (long long)NROI * NC * 4;     // 207360
  const long long O_RPNB = O_RPNC + (long long)Bn * 6 * HW;       // 214272

  // --- bump allocator over workspace ---
  char* wsp = (char*)d_ws;
  auto alloc = [&](size_t bytes) -> void* {
    void* p = (void*)wsp;
    wsp += (bytes + 255) & ~(size_t)255;
    return p;
  };

  auto gemm = [&](const void* A, const void* Bmat, const float* bias, float* Dp,
                  int Mi, int Ni, int Ki, float alpha, int relu, int nb,
                  long long sa, long long sb, long long sd) {
    dim3 grid(cdiv_i(Ni, BN), cdiv_i(Mi, BM), nb);
    edt_gemm_bf16<<<grid, 128, 0, stream>>>((const __bf16*)A, (const __bf16*)Bmat,
                                            bias, Dp, Mi, Ni, Ki, alpha, relu, sa, sb, sd);
  };
  // f32 -> bf16 straight copy; with B column-major this also packs every
  // weight W[O,I] (B_cm[n=o][k=i] == W's own row-major layout).
  auto pack = [&](const float* x, __bf16* y, int n) {
    edt_pack_rows<<<cdiv_i(n, 256), 256, 0, stream>>>(x, y, n);
  };

  // ================= ST on p4 (only live branch) =================
  __bf16* xrows = (__bf16*)alloc((size_t)M * C * 2);
  edt_nchw_to_rows<<<cdiv_i(M * C, 256), 256, 0, stream>>>(feat4, xrows, Bn, C, HW);
  __bf16* pinB = (__bf16*)alloc((size_t)C * C * 2);
  pack(pin_w, pinB, C * C);
  float* x0 = (float*)alloc((size_t)M * C * 4);
  gemm(xrows, pinB, pin_b, x0, M, C, C, 1.f, 0, 1, 0, 0, 0);

  float* cur = x0;
  __bf16* curb = (__bf16*)alloc((size_t)M * C * 2);
  pack(x0, curb, M * C);

  for (int l = 0; l < 2; ++l) {
    const float *in_w = Lw[l][0], *in_b = Lw[l][1], *out_w = Lw[l][2], *out_b = Lw[l][3];
    const float *ln1_g = Lw[l][4], *ln1_b = Lw[l][5];
    const float *w1 = Lw[l][6], *b1 = Lw[l][7], *w2 = Lw[l][8], *b2 = Lw[l][9];
    const float *ln2_g = Lw[l][10], *ln2_b = Lw[l][11];

    // QKV projection
    __bf16* WinB = (__bf16*)alloc((size_t)336 * 112 * 2);
    pack(in_w, WinB, 336 * 112);
    float* qkv = (float*)alloc((size_t)M * 336 * 4);
    gemm(curb, WinB, in_b, qkv, M, 336, 112, 1.f, 0, 1, 0, 0, 0);

    // head split, hd=14 zero-padded to K=16
    size_t asz = (size_t)16 * S * 16;
    __bf16* Qp = (__bf16*)alloc(asz * 2);
    __bf16* Kp = (__bf16*)alloc(asz * 2);
    __bf16* Vt = (__bf16*)alloc(asz * 2);
    edt_pack_qkv<<<cdiv_i(2 * 8 * S * 16, 256), 256, 0, stream>>>(qkv, Qp, Kp, Vt, S);

    // scores = (Q K^T)/sqrt(hd), batched over B*H = 16
    float* scores = (float*)alloc((size_t)16 * S * S * 4);
    gemm(Qp, Kp, nullptr, scores, S, S, 16, ATT_SCALE, 0, 16,
         (long long)S * 16, (long long)S * 16, (long long)S * S);

    // softmax -> bf16 directly
    __bf16* att = (__bf16*)alloc((size_t)16 * S * S * 2);
    edt_softmax<<<cdiv_i(16 * S, 4), 128, 0, stream>>>(scores, att, 16 * S, S);

    // O = att @ V
    float* Ov = (float*)alloc((size_t)16 * S * 16 * 4);
    gemm(att, Vt, nullptr, Ov, S, 16, S, 1.f, 0, 16,
         (long long)S * S, (long long)16 * S, (long long)S * 16);

    // merge heads + output projection
    __bf16* ob = (__bf16*)alloc((size_t)M * C * 2);
    edt_merge_heads<<<cdiv_i(2 * 8 * S * 14, 256), 256, 0, stream>>>(Ov, ob, S);
    __bf16* WoB = (__bf16*)alloc((size_t)C * C * 2);
    pack(out_w, WoB, C * C);
    float* attn = (float*)alloc((size_t)M * C * 4);
    gemm(ob, WoB, out_b, attn, M, C, C, 1.f, 0, 1, 0, 0, 0);

    // x1 = LN1(x + attn)  (fused residual + repack)
    float*  x1  = (float*)alloc((size_t)M * C * 4);
    __bf16* x1b = (__bf16*)alloc((size_t)M * C * 2);
    edt_ln<<<cdiv_i(M, 4), 128, 0, stream>>>(cur, attn, ln1_g, ln1_b, x1, x1b, M, C);

    // FFN
    __bf16* W1B = (__bf16*)alloc((size_t)2048 * 112 * 2);
    pack(w1, W1B, 2048 * 112);
    float* f1 = (float*)alloc((size_t)M * 2048 * 4);
    gemm(x1b, W1B, b1, f1, M, 2048, 112, 1.f, 1, 1, 0, 0, 0);   // fused ReLU
    __bf16* f1b = (__bf16*)alloc((size_t)M * 2048 * 2);
    pack(f1, f1b, M * 2048);
    __bf16* W2B = (__bf16*)alloc((size_t)112 * 2048 * 2);
    pack(w2, W2B, 112 * 2048);
    float* f2 = (float*)alloc((size_t)M * C * 4);
    gemm(f1b, W2B, b2, f2, M, C, 2048, 1.f, 0, 1, 0, 0, 0);

    // x2 = LN2(x1 + f2)
    float*  x2  = (float*)alloc((size_t)M * C * 4);
    __bf16* x2b = (__bf16*)alloc((size_t)M * C * 2);
    edt_ln<<<cdiv_i(M, 4), 128, 0, stream>>>(x1, f2, ln2_g, ln2_b, x2, x2b, M, C);
    cur = x2; curb = x2b;
  }

  // pout projection + residual with original feature -> enh4 (NCHW)
  __bf16* poutB = (__bf16*)alloc((size_t)C * C * 2);
  pack(pout_w, poutB, C * C);
  float* yout = (float*)alloc((size_t)M * C * 4);
  gemm(curb, poutB, pout_b, yout, M, C, C, 1.f, 0, 1, 0, 0, 0);
  float* enh4 = (float*)alloc((size_t)Bn * C * HW * 4);
  edt_add_residual<<<cdiv_i(Bn * C * HW, 256), 256, 0, stream>>>(feat4, yout, enh4, Bn, C, HW);

  // ================= RPN (3x3 conv as im2col GEMM) =================
  __bf16* col = (__bf16*)alloc((size_t)M * 1008 * 2);
  edt_im2col<<<cdiv_i(M * 1008, 256), 256, 0, stream>>>(enh4, col, Bn, C, H4, W4);
  __bf16* c1B = (__bf16*)alloc((size_t)256 * 1008 * 2);
  pack(c1_w, c1B, 256 * 1008);
  float* r = (float*)alloc((size_t)M * 256 * 4);
  gemm(col, c1B, c1_b, r, M, 256, 1008, 1.f, 1, 1, 0, 0, 0);    // fused ReLU
  __bf16* rb = (__bf16*)alloc((size_t)M * 256 * 2);
  pack(r, rb, M * 256);

  __bf16* rclsB = (__bf16*)alloc((size_t)6 * 256 * 2);
  pack(rcls_w, rclsB, 6 * 256);
  float* clsrows = (float*)alloc((size_t)M * 6 * 4);
  gemm(rb, rclsB, rcls_b, clsrows, M, 6, 256, 1.f, 0, 1, 0, 0, 0);
  edt_rows_to_nchw<<<cdiv_i(Bn * 6 * HW, 256), 256, 0, stream>>>(clsrows, out + O_RPNC, Bn, 6, HW);

  __bf16* rbbB = (__bf16*)alloc((size_t)12 * 256 * 2);
  pack(rbb_w, rbbB, 12 * 256);
  float* bbrows = (float*)alloc((size_t)M * 12 * 4);
  gemm(rb, rbbB, rbb_b, bbrows, M, 12, 256, 1.f, 0, 1, 0, 0, 0);
  edt_rows_to_nchw<<<cdiv_i(Bn * 12 * HW, 256), 256, 0, stream>>>(bbrows, out + O_RPNB, Bn, 12, HW);

  // ================= RoIPool + box head =================
  float* roi = (float*)alloc((size_t)NROI * C * 49 * 4);
  edt_roipool<<<NROI, 256, 0, stream>>>(enh4, proposals, roi, C, H4, W4);
  __bf16* roib = (__bf16*)alloc((size_t)NROI * C * 49 * 2);
  pack(roi, roib, NROI * C * 49);

  __bf16* fc1B = (__bf16*)alloc((size_t)1024 * 5488 * 2);
  pack(fc1_w, fc1B, 1024 * 5488);
  float* h1 = (float*)alloc((size_t)NROI * 1024 * 4);
  gemm(roib, fc1B, fc1_b, h1, NROI, 1024, 5488, 1.f, 1, 1, 0, 0, 0);
  __bf16* h1b = (__bf16*)alloc((size_t)NROI * 1024 * 2);
  pack(h1, h1b, NROI * 1024);

  __bf16* fc2B = (__bf16*)alloc((size_t)1024 * 1024 * 2);
  pack(fc2_w, fc2B, 1024 * 1024);
  float* h2 = (float*)alloc((size_t)NROI * 1024 * 4);
  gemm(h1b, fc2B, fc2_b, h2, NROI, 1024, 1024, 1.f, 1, 1, 0, 0, 0);
  __bf16* h2b = (__bf16*)alloc((size_t)NROI * 1024 * 2);
  pack(h2, h2b, NROI * 1024);

  __bf16* hclsB = (__bf16*)alloc((size_t)NC * 1024 * 2);
  pack(hcls_w, hclsB, NC * 1024);
  gemm(h2b, hclsB, hcls_b, out + O_CLS, NROI, NC, 1024, 1.f, 0, 1, 0, 0, 0);

  __bf16* hbbB = (__bf16*)alloc((size_t)NC * 4 * 1024 * 2);
  pack(hbb_w, hbbB, NC * 4 * 1024);
  gemm(h2b, hbbB, hbb_b, out + O_BBOX, NROI, NC * 4, 1024, 1.f, 0, 1, 0, 0, 0);
}
